// CausalSelfAttention_5626407158249
// MI455X (gfx1250) — matrix operand order repaired and checked by reading, not compiled, not run on hardware
//
#include <hip/hip_runtime.h>
#include <hip/hip_bf16.h>
#include <stdint.h>

// ---------------------------------------------------------------------------
// Causal self-attention forward for MI455X (gfx1250).
// - All GEMMs (QKV, QK^T, AV, proj) on v_wmma_f32_16x16x32_bf16 (wave32).
// - GEMM tiles staged by the Tensor Data Mover (tensor_load_to_lds, TENSORcnt).
// - Attention K/V tiles staged via GLOBAL_LOAD_ASYNC_TO_LDS_B128 (ASYNCcnt).
// Both paths double-buffered.
// ---------------------------------------------------------------------------

#define T_SEQ 4096
#define C_DIM 1024
#define H_NUM 16
#define HD    64

typedef unsigned short u16;
typedef unsigned int   u32;
typedef __attribute__((ext_vector_type(16))) __bf16 v16bf;
typedef __attribute__((ext_vector_type(8)))  float  v8f;
typedef __attribute__((ext_vector_type(4)))  u32    u32x4;
typedef __attribute__((ext_vector_type(8)))  u32    u32x8;

union OpBF { uint4 q[2]; v16bf v; };

__device__ __forceinline__ u16 f2bf(float f) {
  union { float f; u32 u; } x; x.f = f;
  u32 r = x.u + 0x7FFFu + ((x.u >> 16) & 1u);   // round-to-nearest-even
  return (u16)(r >> 16);
}

// LDS offset of a __shared__-derived generic pointer (flat LDS aperture:
// addr[31:0] is the LDS byte offset).
__device__ __forceinline__ u32 lds_off(const void* p) {
  return (u32)(uintptr_t)p;
}

// 16-byte async copy global -> LDS (ASYNCcnt-tracked, no VGPR data).
__device__ __forceinline__ void async_g2l_b128(u32 lds_addr, const void* g) {
  asm volatile("global_load_async_to_lds_b128 %0, %1, off"
               :: "v"(lds_addr), "v"((unsigned long long)(uintptr_t)g)
               : "memory");
}

__device__ __forceinline__ void wait_async_le4() {
  asm volatile("s_wait_asynccnt 0x4" ::: "memory");
}
__device__ __forceinline__ void wait_async_0() {
  asm volatile("s_wait_asynccnt 0x0" ::: "memory");
}

// Tensor Data Mover: 2-D tile load (global -> LDS), descriptor per
// CDNA5 ISA. data_size = 2 bytes (bf16); dims/strides
// in elements. Tile rows are packed contiguously at lds_addr.
__device__ __forceinline__ void tdm_load_2d(u32 ldsAddr, const void* gaddr,
                                            u32 tensorX, u32 tensorY,
                                            u32 strideX, u32 tileX, u32 tileY) {
  unsigned long long ga = (unsigned long long)(uintptr_t)gaddr;
  u32 galo = __builtin_amdgcn_readfirstlane((u32)ga);
  u32 gahi = __builtin_amdgcn_readfirstlane((u32)(ga >> 32));
  u32 lad  = __builtin_amdgcn_readfirstlane(ldsAddr);
  u32x4 g0;
  g0[0] = 1u;                                    // count=1, user descriptor
  g0[1] = lad;                                   // lds_addr
  g0[2] = galo;                                  // global_addr[31:0]
  g0[3] = (gahi & 0x01FFFFFFu) | (2u << 30);     // global_addr[56:32] | type=2
  u32x8 g1;
  g1[0] = 1u << 16;                              // data_size=1 (2 bytes)
  g1[1] = (tensorX & 0xFFFFu) << 16;             // tensor_dim0[15:0]
  g1[2] = (tensorX >> 16) | ((tensorY & 0xFFFFu) << 16);   // dim0 hi | dim1 lo
  g1[3] = (tensorY >> 16) | ((tileX & 0xFFFFu) << 16);     // dim1 hi | tile0
  g1[4] = tileY & 0xFFFFu;                       // tile_dim1 (tile_dim2=0)
  g1[5] = strideX;                               // tensor_dim0_stride[31:0]
  g1[6] = 0u;
  g1[7] = 0u;
  asm volatile("tensor_load_to_lds %0, %1" :: "s"(g0), "s"(g1) : "memory");
}

// --------------------------- conversions -----------------------------------

__global__ void k_f32_to_bf16_v4(const float* __restrict__ src,
                                 u16* __restrict__ dst, int n4) {
  int i = blockIdx.x * blockDim.x + threadIdx.x;
  if (i >= n4) return;
  float4 f = ((const float4*)src)[i];
  ushort4 o;
  o.x = f2bf(f.x); o.y = f2bf(f.y); o.z = f2bf(f.z); o.w = f2bf(f.w);
  ((ushort4*)dst)[i] = o;
}

// Wt[n*K + k] = bf16(W[k*N + n])
__global__ void k_transpose_bf16(const float* __restrict__ W,
                                 u16* __restrict__ Wt, int K, int N) {
  long i = (long)blockIdx.x * blockDim.x + threadIdx.x;
  if (i >= (long)K * N) return;
  int n = (int)(i / K), k = (int)(i % K);
  Wt[i] = f2bf(W[(long)k * N + n]);
}

// vt[(h*HD + c)*T + t] = qkv_bf16[t*3C + 2C + h*HD + c]
__global__ void k_build_vt(const u16* __restrict__ qkv, u16* __restrict__ vt) {
  int i = blockIdx.x * blockDim.x + threadIdx.x;    // over C_DIM*T_SEQ
  int t  = i & (T_SEQ - 1);
  int hc = i >> 12;                                 // log2(T_SEQ)
  vt[i] = qkv[(long)t * (3 * C_DIM) + 2 * C_DIM + hc];
}

// --------------------------- NT GEMM ---------------------------------------
// C[M,N] = A[M,K] * Bt[N,K]^T + bias[N]; A,Bt bf16 row-major, f32 accumulate.
// 256 threads = 8 waves (4 M x 2 N); block tile 128x128, K-step 32.
// Tiles staged by TDM (wave 0 issues two 32x128 descriptors per K-step),
// double-buffered on TENSORcnt.

template <bool OUT_F32>
__global__ __launch_bounds__(256)
void k_gemm_nt(const u16* __restrict__ A, const u16* __restrict__ Bt,
               const float* __restrict__ bias, void* __restrict__ Cout,
               int M, int N, int K) {
  __shared__ __align__(16) u16 As[2][128 * 32];
  __shared__ __align__(16) u16 Bs[2][128 * 32];

  const int tid  = threadIdx.x;
  const int lane = tid & 31;
  const int wave = tid >> 5;
  const int r  = lane & 15;
  const int hf = lane >> 4;
  const int wm = wave & 3;       // 0..3 -> M
  const int wn = wave >> 2;      // 0..1 -> N
  const int mBase = blockIdx.y * 128;
  const int nBase = blockIdx.x * 128;

  // One TDM descriptor per 128x32 tile: tile_dim0=32 (K, contiguous),
  // tile_dim1=128 (rows), row stride = K elements.
  auto issue = [&](int kt, int buf) {
    const int kBase = kt << 5;
    tdm_load_2d(lds_off(&As[buf][0]), A + (long)mBase * K + kBase,
                (u32)(K - kBase), (u32)(M - mBase), (u32)K, 32u, 128u);
    tdm_load_2d(lds_off(&Bs[buf][0]), Bt + (long)nBase * K + kBase,
                (u32)(K - kBase), (u32)(N - nBase), (u32)K, 32u, 128u);
  };

  v8f acc[2][4];
#pragma unroll
  for (int mi = 0; mi < 2; ++mi)
#pragma unroll
    for (int ni = 0; ni < 4; ++ni)
#pragma unroll
      for (int v = 0; v < 8; ++v) acc[mi][ni][v] = 0.0f;

  const int nk = K >> 5;
  if (wave == 0) issue(0, 0);
  for (int kt = 0; kt < nk; ++kt) {
    if (wave == 0) {
      if (kt + 1 < nk) {
        issue(kt + 1, (kt + 1) & 1);
        __builtin_amdgcn_s_wait_tensorcnt(2);   // batch kt done; kt+1 in flight
      } else {
        __builtin_amdgcn_s_wait_tensorcnt(0);
      }
    }
    __syncthreads();

    const u16* Ab = As[kt & 1];
    const u16* Bb = Bs[kt & 1];
    OpBF a[2], b[4];
#pragma unroll
    for (int mi = 0; mi < 2; ++mi) {
      const u16* p = Ab + (wm * 32 + mi * 16 + r) * 32;
      a[mi].q[0] = *(const uint4*)(p + hf * 8);          // k = hf*8..+8
      a[mi].q[1] = *(const uint4*)(p + 16 + hf * 8);     // k = 16+hf*8..+8
    }
#pragma unroll
    for (int ni = 0; ni < 4; ++ni) {
      const u16* p = Bb + (wn * 64 + ni * 16 + r) * 32 + hf * 8;  // k = hf*8..+8, 16+hf*8..+8
      b[ni].q[0] = *(const uint4*)(p);
      b[ni].q[1] = *(const uint4*)(p + 16);
    }
#pragma unroll
    for (int mi = 0; mi < 2; ++mi)
#pragma unroll
      for (int ni = 0; ni < 4; ++ni)
        acc[mi][ni] = __builtin_amdgcn_wmma_f32_16x16x32_bf16(
            false, a[mi].v, false, b[ni].v, (short)0, acc[mi][ni], false, false);
    __syncthreads();
  }

  // Epilogue: bias + store. D layout: element v in lane -> M = v + 8*hf, N = r.
#pragma unroll
  for (int ni = 0; ni < 4; ++ni) {
    int col = nBase + wn * 64 + ni * 16 + r;
    float bv = bias[col];
#pragma unroll
    for (int mi = 0; mi < 2; ++mi) {
      int row0 = mBase + wm * 32 + mi * 16 + 8 * hf;
#pragma unroll
      for (int v = 0; v < 8; ++v) {
        float val = acc[mi][ni][v] + bv;
        if (OUT_F32) ((float*)Cout)[(long)(row0 + v) * N + col] = val;
        else         ((u16*)Cout)[(long)(row0 + v) * N + col]   = f2bf(val);
      }
    }
  }
}

// --------------------------- flash attention -------------------------------
// 128 threads = 4 waves; each wave owns 16 query rows, WG owns 64 rows.
// K/V tiles for each 32-key block are cooperatively async-staged into
// double-buffered LDS and shared by all 4 waves. All waves run the
// WG-uniform block count; causal masking neutralizes surplus blocks.

__global__ __launch_bounds__(128)
void k_attn(const u16* __restrict__ qkv, const u16* __restrict__ vt,
            u16* __restrict__ yb) {
  __shared__ __align__(16) u16 Ks[2][32 * 64];    // [key][hd]   4 KB each
  __shared__ __align__(16) u16 Vs[2][64 * 32];    // [hd][key]   4 KB each
  __shared__ __align__(16) u16 plds[4][16 * 32];  // per-wave P scratch

  const int lane = threadIdx.x & 31;
  const int wave = threadIdx.x >> 5;
  const int r  = lane & 15;
  const int hf = lane >> 4;
  const int h  = blockIdx.y;
  const int q0 = blockIdx.x * 64 + wave * 16;

  const u16* Q  = qkv + h * HD;               // row pitch 3C
  const u16* Kp = qkv + C_DIM + h * HD;       // row pitch 3C
  const u16* Vh = vt + (long)h * HD * T_SEQ;  // [c][t], pitch T

  // Stage K tile (32x64, 256 uint4) + V tile (64x32, 256 uint4): 2+2 per thread.
  auto issueKV = [&](int b, int buf) {
    const int kb = b << 5;
#pragma unroll
    for (int j = 0; j < 2; ++j) {
      int idx = (int)threadIdx.x + j * 128;   // 0..255
      int kr = idx >> 3, seg = idx & 7;       // K: 8 uint4 per key row
      async_g2l_b128(lds_off(&Ks[buf][kr * 64 + seg * 8]),
                     Kp + (long)(kb + kr) * (3 * C_DIM) + seg * 8);
      int c = idx >> 2, vseg = idx & 3;       // V: 4 uint4 per channel row
      async_g2l_b128(lds_off(&Vs[buf][c * 32 + vseg * 8]),
                     Vh + (long)c * T_SEQ + kb + vseg * 8);
    }
  };

  // Q operand registers (reused across all key blocks): hd 0..31 / 32..63.
  OpBF aq[2];
  {
    const u16* p = Q + (long)(q0 + r) * (3 * C_DIM);
#pragma unroll
    for (int s = 0; s < 2; ++s) {
      aq[s].q[0] = *(const uint4*)(p + s * 32 + hf * 8);
      aq[s].q[1] = *(const uint4*)(p + s * 32 + 16 + hf * 8);
    }
  }

  v8f o[4];
  float rm[8], rl[8];
#pragma unroll
  for (int c = 0; c < 4; ++c)
#pragma unroll
    for (int v = 0; v < 8; ++v) o[c][v] = 0.0f;
#pragma unroll
  for (int v = 0; v < 8; ++v) { rm[v] = -3.0e38f; rl[v] = 0.0f; }

  const int nb = (blockIdx.x + 1) * 2;   // WG-uniform causal block count
  issueKV(0, 0);
  for (int b = 0; b < nb; ++b) {
    const int kb = b << 5;
    if (b + 1 < nb) {
      issueKV(b + 1, (b + 1) & 1);
      wait_async_le4();
    } else {
      wait_async_0();
    }
    __syncthreads();

    const u16* Kt  = Ks[b & 1];
    const u16* Vtl = Vs[b & 1];

    v8f s0, s1;
#pragma unroll
    for (int v = 0; v < 8; ++v) { s0[v] = 0.0f; s1[v] = 0.0f; }

    // Scores: S[16q x 32k] = Q(16x64) * K(32keys x 64)^T, two hd-steps of 32.
#pragma unroll
    for (int s = 0; s < 2; ++s) {
      OpBF bk0, bk1;
      const u16* p0 = Kt + (r)      * 64 + s * 32 + hf * 8;
      const u16* p1 = Kt + (16 + r) * 64 + s * 32 + hf * 8;
      bk0.q[0] = *(const uint4*)(p0); bk0.q[1] = *(const uint4*)(p0 + 16);
      bk1.q[0] = *(const uint4*)(p1); bk1.q[1] = *(const uint4*)(p1 + 16);
      s0 = __builtin_amdgcn_wmma_f32_16x16x32_bf16(false, aq[s].v, false, bk0.v,
                                                   (short)0, s0, false, false);
      s1 = __builtin_amdgcn_wmma_f32_16x16x32_bf16(false, aq[s].v, false, bk1.v,
                                                   (short)0, s1, false, false);
    }

    // Online softmax update; rows live per (v, hf) across 16 lanes of a half.
    const float scale = 0.125f;   // 1/sqrt(64)
#pragma unroll
    for (int v = 0; v < 8; ++v) {
      int qg = q0 + v + 8 * hf;
      float x0 = (kb + r      <= qg) ? s0[v] * scale : -3.0e38f;
      float x1 = (kb + 16 + r <= qg) ? s1[v] * scale : -3.0e38f;
      float t = fmaxf(x0, x1);
#pragma unroll
      for (int off = 8; off; off >>= 1) t = fmaxf(t, __shfl_xor(t, off, 16));
      float mn = fmaxf(rm[v], t);
      float p0 = __expf(x0 - mn);
      float p1 = __expf(x1 - mn);
      float rs = p0 + p1;
#pragma unroll
      for (int off = 8; off; off >>= 1) rs += __shfl_xor(rs, off, 16);
      float al = __expf(rm[v] - mn);
      rl[v] = rl[v] * al + rs;
      rm[v] = mn;
#pragma unroll
      for (int c = 0; c < 4; ++c) o[c][v] *= al;
      int row = v + 8 * hf;
      plds[wave][row * 32 + r]      = f2bf(p0);
      plds[wave][row * 32 + 16 + r] = f2bf(p1);
    }
    // per-wave LDS roundtrip: ensure P stores landed before operand reload
    asm volatile("s_wait_dscnt 0x0" ::: "memory");

    OpBF ap;
    const u16* pp = &plds[wave][r * 32];
    ap.q[0] = *(const uint4*)(pp + hf * 8);
    ap.q[1] = *(const uint4*)(pp + 16 + hf * 8);

    // O += P(16x32keys) * V(32keys x 64); Vs is [hd][key] so NT-contiguous.
#pragma unroll
    for (int nc = 0; nc < 4; ++nc) {
      OpBF bv;
      const u16* pv = Vtl + (nc * 16 + r) * 32 + hf * 8;
      bv.q[0] = *(const uint4*)(pv);
      bv.q[1] = *(const uint4*)(pv + 16);
      o[nc] = __builtin_amdgcn_wmma_f32_16x16x32_bf16(false, ap.v, false, bv.v,
                                                      (short)0, o[nc], false, false);
    }
    __syncthreads();   // done reading buf[b&1] before it is refilled
  }

  // Normalize and store y (bf16 feeds the projection GEMM).
#pragma unroll
  for (int v = 0; v < 8; ++v) {
    float inv = 1.0f / rl[v];
    int qg = q0 + v + 8 * hf;
#pragma unroll
    for (int nc = 0; nc < 4; ++nc)
      yb[(long)qg * C_DIM + h * HD + nc * 16 + r] = f2bf(o[nc][v] * inv);
  }
}

// --------------------------- launch ----------------------------------------

extern "C" void kernel_launch(void* const* d_in, const int* in_sizes, int n_in,
                              void* d_out, int out_size, void* d_ws, size_t ws_size,
                              hipStream_t stream) {
  const float* x      = (const float*)d_in[0];
  const float* W_attn = (const float*)d_in[1];
  const float* b_attn = (const float*)d_in[2];
  const float* W_proj = (const float*)d_in[3];
  const float* b_proj = (const float*)d_in[4];
  float* out = (float*)d_out;

  // Workspace layout (56 MB total):
  char* ws = (char*)d_ws;
  u16* xb     = (u16*)(ws);                      //  8 MB  x bf16 [T,C]
  u16* wattnT = (u16*)(ws + (8u  << 20));        //  6 MB  W_attn^T bf16 [3C,C]
  u16* wprojT = (u16*)(ws + (14u << 20));        //  2 MB  W_proj^T bf16 [C,C]
  u16* qkvb   = (u16*)(ws + (16u << 20));        // 24 MB  qkv bf16 [T,3C]
  u16* vtb    = (u16*)(ws + (40u << 20));        //  8 MB  V^T bf16 [H*hd,T]
  u16* yb     = (u16*)(ws + (48u << 20));        //  8 MB  y bf16 [T,C]

  // 1) activations + weights to bf16 (weights transposed for NT GEMM)
  k_f32_to_bf16_v4<<<(T_SEQ * C_DIM / 4 + 255) / 256, 256, 0, stream>>>(
      x, xb, T_SEQ * C_DIM / 4);
  k_transpose_bf16<<<(C_DIM * 3 * C_DIM + 255) / 256, 256, 0, stream>>>(
      W_attn, wattnT, C_DIM, 3 * C_DIM);
  k_transpose_bf16<<<(C_DIM * C_DIM + 255) / 256, 256, 0, stream>>>(
      W_proj, wprojT, C_DIM, C_DIM);

  // 2) qkv = x @ W_attn + b_attn      (M=4096, N=3072, K=1024)
  k_gemm_nt<false><<<dim3(3 * C_DIM / 128, T_SEQ / 128), 256, 0, stream>>>(
      xb, wattnT, b_attn, qkvb, T_SEQ, 3 * C_DIM, C_DIM);

  // 3) V^T per head for NT-contiguous AV
  k_build_vt<<<C_DIM * T_SEQ / 256, 256, 0, stream>>>(qkvb, vtb);

  // 4) flash attention (heads x query-tiles)
  k_attn<<<dim3(T_SEQ / 64, H_NUM), 128, 0, stream>>>(qkvb, vtb, yb);

  // 5) out = y @ W_proj + b_proj      (M=4096, N=1024, K=1024), f32 output
  k_gemm_nt<true><<<dim3(C_DIM / 128, T_SEQ / 128), 256, 0, stream>>>(
      yb, wprojT, b_proj, (void*)out, T_SEQ, C_DIM, C_DIM);
}
